// GroupedQueryAttention_52527450030402
// MI455X (gfx1250) — compile-verified, matrix operand order checked by reading, not run on hardware
//
#include <hip/hip_runtime.h>

#define B_   16
#define T_   512
#define HID_ 1024
#define NH_  8
#define NKV_ 4
#define D_   128
#define M_   (B_ * T_)                 // 8192 rows
#define NQKV_ (NH_*D_ + 2*NKV_*D_)     // 2048 fused QKV cols

typedef __attribute__((ext_vector_type(16))) __bf16 v16bf;
typedef __attribute__((ext_vector_type(8)))  float  v8f;

union Frag16 { v16bf v; unsigned int u[8]; };

// K-offset of 32-bit word j inside a 16-bit A/B fragment (ISA 7.12.2):
// words 0..3 -> K = 2j (+8 for upper half-wave), words 4..7 -> K = 16+2(j-4).
// Words 0..3 and 4..7 are contiguous 16B runs -> compiler emits 2x b128 loads.
__device__ __forceinline__ int koff(int j) { return (j < 4) ? (2 * j) : (8 + 2 * j); }

__device__ __forceinline__ unsigned short f2bf(float x) {
  unsigned int u = __float_as_uint(x);
  u += 0x7FFFu + ((u >> 16) & 1u);          // round-to-nearest-even
  return (unsigned short)(u >> 16);
}

__device__ __forceinline__ float bf2f(unsigned short u) {
  return __uint_as_float((unsigned int)u << 16);
}

__device__ __forceinline__ v8f vzero8() {
  v8f v;
#pragma unroll
  for (int i = 0; i < 8; ++i) v[i] = 0.0f;
  return v;
}

// ---------------------------------------------------------------------------
// Elementwise converters
// ---------------------------------------------------------------------------
__global__ void cvt_f32_bf16(const float* __restrict__ src,
                             unsigned short* __restrict__ dst, int n) {
  int i = blockIdx.x * blockDim.x + threadIdx.x;
  if (i < n) dst[i] = f2bf(src[i]);
}

// src: K x N (row-major) f32  ->  dst: N x K (row-major) bf16
__global__ void cvt_transpose_bf16(const float* __restrict__ src,
                                   unsigned short* __restrict__ dst,
                                   int K, int N) {
  int i = blockIdx.x * blockDim.x + threadIdx.x;
  if (i < K * N) {
    int k = i / N, n = i % N;
    dst[(size_t)n * K + k] = f2bf(src[i]);
  }
}

// ---------------------------------------------------------------------------
// bf16 WMMA GEMM: C[M,N] = A[M,K] * B[K,N]   (B given transposed: Bt[N,K])
// block = 256 thr = 8 waves laid out 4(M) x 2(N); block tile 128 x 128;
// wave tile 32(M) x 64(N): 2 A-frags x 4 B-frags -> 8 WMMAs per K-chunk.
// Epilogue: f32 store (Cf) or bf16 store (Cb) selected by non-null pointer.
// ---------------------------------------------------------------------------
__global__ __launch_bounds__(256) void gemm_bf16_wmma(
    const unsigned short* __restrict__ A,
    const unsigned short* __restrict__ Bt,
    float* __restrict__ Cf,
    unsigned short* __restrict__ Cb,
    int M, int N, int K) {
  const int wave = threadIdx.x >> 5;
  const int lane = threadIdx.x & 31;
  const int hi8  = (lane >> 4) << 3;     // upper half-wave K (and M) shift
  const int l16  = lane & 15;

  const int mBase = blockIdx.y * 128 + (wave >> 1) * 32;
  const int nBase = blockIdx.x * 128 + (wave & 1) * 64;

  v8f acc[2][4];
#pragma unroll
  for (int s = 0; s < 2; ++s)
#pragma unroll
    for (int n = 0; n < 4; ++n) acc[s][n] = vzero8();

  const unsigned short* arow0 = A  + (size_t)(mBase +  0 + l16) * K;
  const unsigned short* arow1 = A  + (size_t)(mBase + 16 + l16) * K;
  const unsigned short* brow0 = Bt + (size_t)(nBase +  0 + l16) * K;
  const unsigned short* brow1 = Bt + (size_t)(nBase + 16 + l16) * K;
  const unsigned short* brow2 = Bt + (size_t)(nBase + 32 + l16) * K;
  const unsigned short* brow3 = Bt + (size_t)(nBase + 48 + l16) * K;

  for (int k0 = 0; k0 < K; k0 += 32) {
    // pull next K-chunk lines toward the WGP while this chunk computes
    __builtin_prefetch(arow0 + k0 + 64, 0, 1);
    __builtin_prefetch(arow1 + k0 + 64, 0, 1);
    __builtin_prefetch(brow0 + k0 + 64, 0, 1);
    __builtin_prefetch(brow1 + k0 + 64, 0, 1);
    __builtin_prefetch(brow2 + k0 + 64, 0, 1);
    __builtin_prefetch(brow3 + k0 + 64, 0, 1);

    Frag16 a0, a1, b0, b1, b2, b3;
#pragma unroll
    for (int j = 0; j < 8; ++j) {
      const int ko = k0 + koff(j) + hi8;
      a0.u[j] = *(const unsigned int*)(arow0 + ko);
      a1.u[j] = *(const unsigned int*)(arow1 + ko);
      b0.u[j] = *(const unsigned int*)(brow0 + ko);
      b1.u[j] = *(const unsigned int*)(brow1 + ko);
      b2.u[j] = *(const unsigned int*)(brow2 + ko);
      b3.u[j] = *(const unsigned int*)(brow3 + ko);
    }
    acc[0][0] = __builtin_amdgcn_wmma_f32_16x16x32_bf16(false, a0.v, false, b0.v, (short)0, acc[0][0], false, false);
    acc[0][1] = __builtin_amdgcn_wmma_f32_16x16x32_bf16(false, a0.v, false, b1.v, (short)0, acc[0][1], false, false);
    acc[0][2] = __builtin_amdgcn_wmma_f32_16x16x32_bf16(false, a0.v, false, b2.v, (short)0, acc[0][2], false, false);
    acc[0][3] = __builtin_amdgcn_wmma_f32_16x16x32_bf16(false, a0.v, false, b3.v, (short)0, acc[0][3], false, false);
    acc[1][0] = __builtin_amdgcn_wmma_f32_16x16x32_bf16(false, a1.v, false, b0.v, (short)0, acc[1][0], false, false);
    acc[1][1] = __builtin_amdgcn_wmma_f32_16x16x32_bf16(false, a1.v, false, b1.v, (short)0, acc[1][1], false, false);
    acc[1][2] = __builtin_amdgcn_wmma_f32_16x16x32_bf16(false, a1.v, false, b2.v, (short)0, acc[1][2], false, false);
    acc[1][3] = __builtin_amdgcn_wmma_f32_16x16x32_bf16(false, a1.v, false, b3.v, (short)0, acc[1][3], false, false);
  }

#pragma unroll
  for (int s = 0; s < 2; ++s) {
#pragma unroll
    for (int r = 0; r < 8; ++r) {
      const size_t row = (size_t)(mBase + s * 16 + hi8 + r) * N + nBase + l16;
      if (Cb) {
#pragma unroll
        for (int n = 0; n < 4; ++n) Cb[row + n * 16] = f2bf(acc[s][n][r]);
      } else {
#pragma unroll
        for (int n = 0; n < 4; ++n) Cf[row + n * 16] = acc[s][n][r];
      }
    }
  }
}

// ---------------------------------------------------------------------------
// RoPE + layout kernels.  QKV buffer: [B*T, 2048] bf16 (Q | K | V)
// ---------------------------------------------------------------------------
__global__ void rope_q_kernel(const unsigned short* __restrict__ QKV,
                              unsigned short* __restrict__ Qr) {  // [B,NH,T,D]
  int i = blockIdx.x * blockDim.x + threadIdx.x;   // B*NH*T*D threads
  const int d = i & (D_ - 1);
  const int t = (i >> 7) & (T_ - 1);
  const int h = (i >> 16) & (NH_ - 1);
  const int b = i >> 19;
  const size_t src = (size_t)(b * T_ + t) * NQKV_ + h * D_;
  const float x  = bf2f(QKV[src + d]);
  const float xr = (d < 64) ? -bf2f(QKV[src + d + 64]) : bf2f(QKV[src + d - 64]);
  const int   dm = d & 63;
  const float freq = __expf(-9.210340371976184f * (float)dm / 64.0f); // theta^(-2dm/D)
  float s, c;
  __sincosf((float)t * freq, &s, &c);
  Qr[i] = f2bf((x * c + xr * s) * 0.08838834764831845f);  // fold 1/sqrt(D)
}

__global__ void rope_k_kernel(const unsigned short* __restrict__ QKV,
                              unsigned short* __restrict__ Kr) {  // [B,NKV,T,D]
  int i = blockIdx.x * blockDim.x + threadIdx.x;   // B*NKV*T*D threads
  const int d  = i & (D_ - 1);
  const int t  = (i >> 7) & (T_ - 1);
  const int kv = (i >> 16) & (NKV_ - 1);
  const int b  = i >> 18;
  const size_t src = (size_t)(b * T_ + t) * NQKV_ + NH_ * D_ + kv * D_;
  const float x  = bf2f(QKV[src + d]);
  const float xr = (d < 64) ? -bf2f(QKV[src + d + 64]) : bf2f(QKV[src + d - 64]);
  const int   dm = d & 63;
  const float freq = __expf(-9.210340371976184f * (float)dm / 64.0f);
  float s, c;
  __sincosf((float)t * freq, &s, &c);
  Kr[i] = f2bf(x * c + xr * s);
}

__global__ void v_transpose_kernel(const unsigned short* __restrict__ QKV,
                                   unsigned short* __restrict__ Vt) { // [B,NKV,D,T]
  int i = blockIdx.x * blockDim.x + threadIdx.x;   // B*NKV*D*T threads
  const int t  = i & (T_ - 1);
  const int d  = (i >> 9) & (D_ - 1);
  const int kv = (i >> 16) & (NKV_ - 1);
  const int b  = i >> 18;
  Vt[i] = QKV[(size_t)(b * T_ + t) * NQKV_ + (NH_ + NKV_) * D_ + kv * D_ + d];
}

// ---------------------------------------------------------------------------
// Causal flash attention. block = 8 waves; wave = 16 queries; key blocks of 32.
// ---------------------------------------------------------------------------
__global__ __launch_bounds__(256) void flash_attn_kernel(
    const unsigned short* __restrict__ Q,   // [B,NH,T,D]
    const unsigned short* __restrict__ Kc,  // [B,NKV,T,D]
    const unsigned short* __restrict__ Vt,  // [B,NKV,D,T]
    unsigned short* __restrict__ O) {       // [B,T,NH*D]
  __shared__ unsigned short plds[8][16][32];   // per-wave P tile (C->A relayout)

  const int wave = threadIdx.x >> 5;
  const int lane = threadIdx.x & 31;
  const int hi8  = (lane >> 4) << 3;
  const int l16  = lane & 15;

  const int bh = blockIdx.y;
  const int b  = bh >> 3;
  const int h  = bh & 7;
  const int kv = h >> 1;                       // NH/NKV = 2
  const int q0 = blockIdx.x * 128 + wave * 16;

  // Q fragments: 4 K-chunks of 32 along D (Q pre-scaled by 1/sqrt(D))
  Frag16 qf[4];
  {
    const unsigned short* qrow = Q + ((size_t)(b * NH_ + h) * T_ + (q0 + l16)) * D_;
#pragma unroll
    for (int c = 0; c < 4; ++c)
#pragma unroll
      for (int j = 0; j < 8; ++j)
        qf[c].u[j] = *(const unsigned int*)(qrow + c * 32 + koff(j) + hi8);
  }

  v8f Of[8];
#pragma unroll
  for (int f = 0; f < 8; ++f) Of[f] = vzero8();
  float rowmax[8], rowsum[8];
#pragma unroll
  for (int r = 0; r < 8; ++r) { rowmax[r] = -3.0e38f; rowsum[r] = 0.0f; }

  const unsigned short* kbase = Kc + (size_t)(b * NKV_ + kv) * T_ * D_;
  const unsigned short* vbase = Vt + (size_t)(b * NKV_ + kv) * D_ * T_;

  const int nkb = ((q0 + 15) >> 5) + 1;        // causal bound
  for (int kb = 0; kb < nkb; ++kb) {
    const int ks = kb << 5;

    // ---- S = Q * K^T for 32 keys (two 16-key halves) ----
    v8f Sa = vzero8(), Sb = vzero8();
#pragma unroll
    for (int c = 0; c < 4; ++c) {
      Frag16 ka, kbf;
      const unsigned short* krA = kbase + (size_t)(ks + l16)      * D_ + c * 32;
      const unsigned short* krB = kbase + (size_t)(ks + 16 + l16) * D_ + c * 32;
#pragma unroll
      for (int j = 0; j < 8; ++j) {
        const int ko = koff(j) + hi8;
        ka.u[j]  = *(const unsigned int*)(krA + ko);
        kbf.u[j] = *(const unsigned int*)(krB + ko);
      }
      Sa = __builtin_amdgcn_wmma_f32_16x16x32_bf16(false, qf[c].v, false, ka.v,  (short)0, Sa, false, false);
      Sb = __builtin_amdgcn_wmma_f32_16x16x32_bf16(false, qf[c].v, false, kbf.v, (short)0, Sb, false, false);
    }

    // ---- causal mask + online softmax (row lives in one 16-lane half) ----
    const int keyA = ks + l16;
    const int keyB = keyA + 16;
#pragma unroll
    for (int r = 0; r < 8; ++r) {
      const int m = q0 + hi8 + r;
      const float sa = (keyA <= m) ? Sa[r] : -3.0e38f;
      const float sb = (keyB <= m) ? Sb[r] : -3.0e38f;
      float mx = fmaxf(sa, sb);
#pragma unroll
      for (int off = 1; off < 16; off <<= 1)
        mx = fmaxf(mx, __shfl_xor(mx, off, 32));
      const float nm    = fmaxf(rowmax[r], mx);
      const float alpha = __expf(rowmax[r] - nm);
      rowmax[r] = nm;
      rowsum[r] *= alpha;
#pragma unroll
      for (int f = 0; f < 8; ++f) Of[f][r] *= alpha;
      const float pa = __expf(sa - nm);
      const float pb = __expf(sb - nm);
      float ps = pa + pb;
#pragma unroll
      for (int off = 1; off < 16; off <<= 1)
        ps += __shfl_xor(ps, off, 32);
      rowsum[r] += ps;
      plds[wave][hi8 + r][l16]      = f2bf(pa);
      plds[wave][hi8 + r][16 + l16] = f2bf(pb);
    }
    __builtin_amdgcn_wave_barrier();   // wave-local LDS: keep store->load order

    // ---- P (A-layout) from LDS, then O += P * V ----
    Frag16 pf;
#pragma unroll
    for (int j = 0; j < 8; ++j)
      pf.u[j] = *(const unsigned int*)&plds[wave][l16][koff(j) + hi8];

#pragma unroll
    for (int f = 0; f < 8; ++f) {
      Frag16 vf;
      const unsigned short* vrow = vbase + (size_t)(f * 16 + l16) * T_ + ks;
#pragma unroll
      for (int j = 0; j < 8; ++j)
        vf.u[j] = *(const unsigned int*)(vrow + koff(j) + hi8);
      Of[f] = __builtin_amdgcn_wmma_f32_16x16x32_bf16(false, pf.v, false, vf.v, (short)0, Of[f], false, false);
    }
    __builtin_amdgcn_wave_barrier();
  }

  // ---- normalize, store bf16 context as [b, t, h*D + d] for the Wo GEMM ----
#pragma unroll
  for (int r = 0; r < 8; ++r) {
    const float inv = 1.0f / rowsum[r];
    unsigned short* orow =
        O + (size_t)(b * T_ + q0 + hi8 + r) * (NH_ * D_) + h * D_ + l16;
#pragma unroll
    for (int f = 0; f < 8; ++f) orow[f * 16] = f2bf(Of[f][r] * inv);
  }
}

// ---------------------------------------------------------------------------
// Launcher
// ---------------------------------------------------------------------------
extern "C" void kernel_launch(void* const* d_in, const int* in_sizes, int n_in,
                              void* d_out, int out_size, void* d_ws, size_t ws_size,
                              hipStream_t stream) {
  const float* X  = (const float*)d_in[0];
  const float* Wq = (const float*)d_in[1];
  const float* Wk = (const float*)d_in[2];
  const float* Wv = (const float*)d_in[3];
  const float* Wo = (const float*)d_in[4];
  float* out = (float*)d_out;

  char* ws = (char*)d_ws;                                  // scratch layout (bytes)
  unsigned short* Xbf   = (unsigned short*)(ws + 0);         // 16 MiB  X bf16
  unsigned short* WqkvT = (unsigned short*)(ws + 16777216);  //  4 MiB  [2048,1024]
  unsigned short* WoT   = (unsigned short*)(ws + 20971520);  //  2 MiB  [1024,1024]
  unsigned short* QKV   = (unsigned short*)(ws + 23068672);  // 32 MiB  [8192,2048] bf16
  unsigned short* Qr    = (unsigned short*)(ws + 56623104);  // 16 MiB  [B,NH,T,D]
  unsigned short* Kr    = (unsigned short*)(ws + 73400320);  //  8 MiB  [B,NKV,T,D]
  unsigned short* Vt    = (unsigned short*)(ws + 81788928);  //  8 MiB  [B,NKV,D,T]
  unsigned short* At    = (unsigned short*)(ws + 90177536);  // 16 MiB  [8192,1024]

  // 1) convert inputs to bf16 (weights transposed so B-frags load contiguously)
  cvt_f32_bf16<<<(M_ * HID_) / 256, 256, 0, stream>>>(X, Xbf, M_ * HID_);
  cvt_transpose_bf16<<<(HID_ * 1024) / 256, 256, 0, stream>>>(Wq, WqkvT,               HID_, 1024);
  cvt_transpose_bf16<<<(HID_ *  512) / 256, 256, 0, stream>>>(Wk, WqkvT + 1024 * HID_, HID_,  512);
  cvt_transpose_bf16<<<(HID_ *  512) / 256, 256, 0, stream>>>(Wv, WqkvT + 1536 * HID_, HID_,  512);
  cvt_transpose_bf16<<<(1024 * HID_) / 256, 256, 0, stream>>>(Wo, WoT,                 1024, HID_);

  // 2) fused QKV projection GEMM (8192 x 2048), bf16 output
  dim3 g1(NQKV_ / 128, M_ / 128);
  gemm_bf16_wmma<<<g1, 256, 0, stream>>>(Xbf, WqkvT, (float*)nullptr, QKV,
                                         M_, NQKV_, HID_);

  // 3) RoPE + attention layouts
  rope_q_kernel<<<(B_ * NH_  * T_ * D_) / 256, 256, 0, stream>>>(QKV, Qr);
  rope_k_kernel<<<(B_ * NKV_ * T_ * D_) / 256, 256, 0, stream>>>(QKV, Kr);
  v_transpose_kernel<<<(B_ * NKV_ * D_ * T_) / 256, 256, 0, stream>>>(QKV, Vt);

  // 4) causal flash attention
  dim3 g2(T_ / 128, B_ * NH_);
  flash_attn_kernel<<<g2, 256, 0, stream>>>(Qr, Kr, Vt, At);

  // 5) output projection GEMM (8192 x 1024) -> f32 out
  dim3 g3(HID_ / 128, M_ / 128);
  gemm_bf16_wmma<<<g3, 256, 0, stream>>>(At, WoT, out, (unsigned short*)nullptr,
                                         M_, HID_, HID_);
}
